// EdgePredModel_36971078484656
// MI455X (gfx1250) — compile-verified, matrix operand order checked by reading, not run on hardware
//
#include <hip/hip_runtime.h>

typedef __attribute__((ext_vector_type(16))) _Float16 v16h;
typedef __attribute__((ext_vector_type(8)))  _Float16 v8h;
typedef __attribute__((ext_vector_type(8)))  float    v8f;

#define THREADS 256

#if defined(__gfx1250__) && __has_builtin(__builtin_amdgcn_global_load_async_to_lds_b128)
#define USE_ASYNC_LDS 1
#else
#define USE_ASYNC_LDS 0
#endif

#if defined(__gfx1250__) && __has_builtin(__builtin_amdgcn_s_wait_asynccnt)
#define WAIT_ASYNC(n) __builtin_amdgcn_s_wait_asynccnt(n)
#elif defined(__gfx1250__)
#define WAIT_ASYNC(n) asm volatile("s_wait_asynccnt %0" ::"i"(n) : "memory")
#else
#define WAIT_ASYNC(n)
#endif

// Exact-match types for the async-LDS builtin:
//   arg0: global (AS1) pointer to 16B int vector; arg1: LDS (AS3) pointer.
typedef int gcc_int4 __attribute__((__vector_size__(16)));
typedef __attribute__((address_space(1))) gcc_int4 as1_int4;
typedef __attribute__((address_space(3))) gcc_int4 as3_int4;

// ---------------------------------------------------------------------------
// Elementwise / conversion kernels
// ---------------------------------------------------------------------------
__global__ void k_f32_to_f16(const float* __restrict__ in, _Float16* __restrict__ out, size_t n) {
  size_t i = (size_t)blockIdx.x * blockDim.x + threadIdx.x;
  size_t st = (size_t)gridDim.x * blockDim.x;
  for (; i < n; i += st) out[i] = (_Float16)in[i];
}

// w: (K,N) row-major fp32  ->  wt: (N,K) row-major f16  (i.e. W^T)
__global__ void k_transpose_to_f16(const float* __restrict__ w, _Float16* __restrict__ wt,
                                   int K, int N) {
  size_t n = (size_t)K * N;
  size_t i = (size_t)blockIdx.x * blockDim.x + threadIdx.x;
  size_t st = (size_t)gridDim.x * blockDim.x;
  for (; i < n; i += st) {
    int nn = (int)(i / K);
    int kk = (int)(i % K);
    wt[i] = (_Float16)w[(size_t)kk * N + nn];
  }
}

__global__ void k_fill_f32(float* __restrict__ p, float v, size_t n) {
  size_t i = (size_t)blockIdx.x * blockDim.x + threadIdx.x;
  size_t st = (size_t)gridDim.x * blockDim.x;
  for (; i < n; i += st) p[i] = v;
}

__global__ void k_fill_u32(unsigned* __restrict__ p, unsigned v, size_t n) {
  size_t i = (size_t)blockIdx.x * blockDim.x + threadIdx.x;
  size_t st = (size_t)gridDim.x * blockDim.x;
  for (; i < n; i += st) p[i] = v;
}

__global__ void k_relu_f32_to_f16(const float* __restrict__ in, _Float16* __restrict__ out, size_t n) {
  size_t i = (size_t)blockIdx.x * blockDim.x + threadIdx.x;
  size_t st = (size_t)gridDim.x * blockDim.x;
  for (; i < n; i += st) out[i] = (_Float16)fmaxf(in[i], 0.0f);
}

// ---------------------------------------------------------------------------
// f16 WMMA GEMM:  C(MxN,f16) = act( A(MxK,f16,row) @ BT(NxK,f16,row)^T + bias )
// Block tile 64(M) x 128(N), K-step 32. 8 waves = 4(M) x 2(N); each wave does
// a 16x64 tile = 4 WMMAs per K-step, reusing one A fragment.
// Staging: async global->LDS (double-buffered, s_wait_asynccnt pipelined) when
// available, else register-prefetch double buffering.
// Requires N % 128 == 0, K % 32 == 0; any M (row-clamped loads, guarded stores).
// ---------------------------------------------------------------------------
__device__ __forceinline__ v16h load_frag(const _Float16* row, int lhalf) {
  // 16-bit fragment: K chunks at halves [8*lhalf, +8) and [16+8*lhalf, +8)
  const v8h c0 = *(const v8h*)(row + lhalf * 8);
  const v8h c1 = *(const v8h*)(row + 16 + lhalf * 8);
  v16h r;
#pragma unroll
  for (int i = 0; i < 8; ++i) { r[i] = c0[i]; r[i + 8] = c1[i]; }
  return r;
}

__global__ void k_gemm_f16(const _Float16* __restrict__ A, const _Float16* __restrict__ BT,
                           const float* __restrict__ bias, _Float16* __restrict__ C,
                           int M, int N, int K, int relu) {
#if USE_ASYNC_LDS
  __shared__ _Float16 As[2][64 * 32];    // 2 x 4KB
  __shared__ _Float16 Bs[2][128 * 32];   // 2 x 8KB
#else
  __shared__ _Float16 As[1][64 * 32];
  __shared__ _Float16 Bs[1][128 * 32];
#endif
  const int mbase = blockIdx.x * 64;
  const int nbase = blockIdx.y * 128;
  const int t     = threadIdx.x;
  const int lane  = t & 31;
  const int wave  = t >> 5;
  const int wm    = wave & 3;   // M sub-tile (0..3)
  const int wn    = wave >> 2;  // N sub-tile (0..1), 64 cols each
  const int lrow  = lane & 15;
  const int lhalf = lane >> 4;

  // cooperative staging indices
  const int arow = t >> 2, acg = t & 3;  // A: 64 rows x 4 groups of 8 halves (16B)
  const int brow = t >> 1, bcg = t & 1;  // B: 128 rows x 2 groups of 16 halves (2x16B)

  const int grow = mbase + arow;
  const size_t aoff = (size_t)(grow < M ? grow : (M - 1)) * K;  // clamp; stores guarded
  const size_t boff = (size_t)(nbase + brow) * K;

  v8f acc[4] = {};

#if USE_ASYNC_LDS
  auto issue = [&](int k0, int buf) {
    __builtin_amdgcn_global_load_async_to_lds_b128(
        (as1_int4*)(A + aoff + k0 + acg * 8),
        (as3_int4*)(&As[buf][arow * 32 + acg * 8]), 0, 0);
    __builtin_amdgcn_global_load_async_to_lds_b128(
        (as1_int4*)(BT + boff + k0 + bcg * 16),
        (as3_int4*)(&Bs[buf][brow * 32 + bcg * 16]), 0, 0);
    __builtin_amdgcn_global_load_async_to_lds_b128(
        (as1_int4*)(BT + boff + k0 + bcg * 16 + 8),
        (as3_int4*)(&Bs[buf][brow * 32 + bcg * 16 + 8]), 0, 0);
  };
  issue(0, 0);
  int cur = 0;
#else
  uint4 ra, rb0, rb1;
  ra  = *(const uint4*)&A[aoff + acg * 8];
  rb0 = *(const uint4*)&BT[boff + bcg * 16];
  rb1 = *(const uint4*)&BT[boff + bcg * 16 + 8];
  const int cur = 0;
#endif

  for (int k0 = 0; k0 < K; k0 += 32) {
    const bool hasNext = (k0 + 32) < K;
#if USE_ASYNC_LDS
    if (hasNext) {
      issue(k0 + 32, cur ^ 1);  // next tile in flight during this tile's MMAs
      WAIT_ASYNC(3);            // we issue 3/tile, in-order: current tile done
    } else {
      WAIT_ASYNC(0);
    }
    __syncthreads();
#else
    __syncthreads();
    *(uint4*)&As[0][arow * 32 + acg * 8] = ra;
    *(uint4*)&Bs[0][brow * 32 + bcg * 16] = rb0;
    *(uint4*)&Bs[0][brow * 32 + bcg * 16 + 8] = rb1;
    if (hasNext) {
      ra  = *(const uint4*)&A[aoff + k0 + 32 + acg * 8];
      rb0 = *(const uint4*)&BT[boff + k0 + 32 + bcg * 16];
      rb1 = *(const uint4*)&BT[boff + k0 + 32 + bcg * 16 + 8];
    }
    __syncthreads();
#endif

    const _Float16* as = &As[cur][0];
    const _Float16* bs = &Bs[cur][0];
    const v16h a = load_frag(as + (wm * 16 + lrow) * 32, lhalf);
#pragma unroll
    for (int j = 0; j < 4; ++j) {
      const v16h b = load_frag(bs + (wn * 64 + j * 16 + lrow) * 32, lhalf);
      acc[j] = __builtin_amdgcn_wmma_f32_16x16x32_f16(false, a, false, b,
                                                      (short)0, acc[j], false, false);
    }
    __syncthreads();
#if USE_ASYNC_LDS
    cur ^= 1;
#endif
  }

  // C/D layout: lane L holds col N = L%16; VGPR r holds row M = r + 8*(L/16)
#pragma unroll
  for (int j = 0; j < 4; ++j) {
    const int col = nbase + wn * 64 + j * 16 + lrow;
    const float bv = bias ? bias[col] : 0.0f;
#pragma unroll
    for (int r = 0; r < 8; ++r) {
      const int row = mbase + wm * 16 + lhalf * 8 + r;
      if (row < M) {
        float v = acc[j][r] + bv;
        if (relu) v = fmaxf(v, 0.0f);
        C[(size_t)row * N + col] = (_Float16)v;
      }
    }
  }
}

// ---------------------------------------------------------------------------
// GATv2 attention pieces
// ---------------------------------------------------------------------------
__device__ __forceinline__ unsigned f32key(float f) {
  unsigned u = __float_as_uint(f);
  return (u & 0x80000000u) ? ~u : (u | 0x80000000u);
}
__device__ __forceinline__ float keyf32(unsigned k) {
  unsigned u = (k & 0x80000000u) ? (k & 0x7fffffffu) : ~k;
  return __uint_as_float(u);
}

// logits[e*H+h] = sum_d attn[h,d] * leaky_relu(fs[src[e],h,d] + fd[dst[e],h,d])
__global__ void k_edge_logits(const _Float16* __restrict__ fs, const _Float16* __restrict__ fd,
                              const int* __restrict__ src, const int* __restrict__ dst,
                              const float* __restrict__ attn, float* __restrict__ logits,
                              int E, int H, int D) {
  const int HD = H * D;
  size_t n = (size_t)E * H;
  size_t i = (size_t)blockIdx.x * blockDim.x + threadIdx.x;
  size_t st = (size_t)gridDim.x * blockDim.x;
  for (; i < n; i += st) {
    int e = (int)(i / H), h = (int)(i % H);
    const _Float16* ps = fs + (size_t)src[e] * HD + h * D;
    const _Float16* pd = fd + (size_t)dst[e] * HD + h * D;
    const float*    pa = attn + h * D;
    float acc = 0.0f;
    for (int d0 = 0; d0 < D; d0 += 8) {
      const v8h s = *(const v8h*)(ps + d0);
      const v8h q = *(const v8h*)(pd + d0);
#pragma unroll
      for (int j = 0; j < 8; ++j) {
        float v = (float)s[j] + (float)q[j];
        v = v > 0.0f ? v : 0.2f * v;
        acc += pa[d0 + j] * v;
      }
    }
    logits[i] = acc;
  }
}

__global__ void k_seg_max(const float* __restrict__ logits, const int* __restrict__ dst,
                          unsigned* __restrict__ maxk, int E, int H) {
  size_t n = (size_t)E * H;
  size_t i = (size_t)blockIdx.x * blockDim.x + threadIdx.x;
  size_t st = (size_t)gridDim.x * blockDim.x;
  for (; i < n; i += st) {
    int e = (int)(i / H), h = (int)(i % H);
    atomicMax(&maxk[(size_t)dst[e] * H + h], f32key(logits[i]));
  }
}

__global__ void k_seg_expsum(float* __restrict__ logits, const int* __restrict__ dst,
                             const unsigned* __restrict__ maxk, float* __restrict__ sums,
                             int E, int H) {
  size_t n = (size_t)E * H;
  size_t i = (size_t)blockIdx.x * blockDim.x + threadIdx.x;
  size_t st = (size_t)gridDim.x * blockDim.x;
  for (; i < n; i += st) {
    int e = (int)(i / H), h = (int)(i % H);
    float m = keyf32(maxk[(size_t)dst[e] * H + h]);
    float w = expf(logits[i] - m);
    logits[i] = w;
    atomicAdd(&sums[(size_t)dst[e] * H + h], w);
  }
}

// out[dst, h, d] += (w_e / sum[dst,h]) * fs[src, h, d]
__global__ void k_aggregate(const float* __restrict__ logits, const float* __restrict__ sums,
                            const _Float16* __restrict__ fs,
                            const int* __restrict__ src, const int* __restrict__ dst,
                            float* __restrict__ out, int E, int H, int D) {
  const int D4 = D / 4, HD = H * D;
  size_t n = (size_t)E * H * D4;
  size_t i = (size_t)blockIdx.x * blockDim.x + threadIdx.x;
  size_t st = (size_t)gridDim.x * blockDim.x;
  for (; i < n; i += st) {
    size_t e = i / ((size_t)H * D4);
    int r = (int)(i % ((size_t)H * D4));
    int h = r / D4, d0 = (r % D4) * 4;
    int sn = src[e], dn = dst[e];
    float coef = logits[e * H + h] / sums[(size_t)dn * H + h];
    const _Float16* ps = fs + (size_t)sn * HD + h * D + d0;
    float* po = out + (size_t)dn * HD + h * D + d0;
#pragma unroll
    for (int j = 0; j < 4; ++j) atomicAdd(&po[j], coef * (float)ps[j]);
  }
}

// ---------------------------------------------------------------------------
// Edge feature gather: F[e] = [h1[src], h1[dst], h2[src], h2[dst]] (768 f16)
// 16-byte chunks: 96 chunks per edge; all section boundaries are multiples of 8.
// ---------------------------------------------------------------------------
__global__ void k_gather_edges(const _Float16* __restrict__ h1, const _Float16* __restrict__ h2,
                               const int* __restrict__ src, const int* __restrict__ dst,
                               _Float16* __restrict__ F, int E) {
  size_t n = (size_t)E * 96;
  size_t i = (size_t)blockIdx.x * blockDim.x + threadIdx.x;
  size_t st = (size_t)gridDim.x * blockDim.x;
  for (; i < n; i += st) {
    size_t e = i / 96;
    int c = (int)(i % 96) * 8;
    const _Float16* p;
    if      (c < 128) p = h1 + (size_t)src[e] * 128 + c;
    else if (c < 256) p = h1 + (size_t)dst[e] * 128 + (c - 128);
    else if (c < 512) p = h2 + (size_t)src[e] * 256 + (c - 256);
    else              p = h2 + (size_t)dst[e] * 256 + (c - 512);
    *(uint4*)&F[e * 768 + c] = *(const uint4*)p;
  }
}

// Final 256 -> 1 layer: one wave per edge, coalesced v8h loads + shuffle reduce.
__global__ void k_final_dot(const _Float16* __restrict__ e1, const float* __restrict__ w2,
                            const float* __restrict__ b2, float* __restrict__ out, int E) {
  const int lane = threadIdx.x & 31;
  const int wpb  = blockDim.x >> 5;
  int e  = blockIdx.x * wpb + (threadIdx.x >> 5);
  const int nw = gridDim.x * wpb;
  for (; e < E; e += nw) {
    const v8h v = *(const v8h*)(e1 + (size_t)e * 256 + lane * 8);
    float acc = 0.0f;
#pragma unroll
    for (int j = 0; j < 8; ++j) acc += (float)v[j] * w2[lane * 8 + j];
#pragma unroll
    for (int off = 16; off > 0; off >>= 1) acc += __shfl_down(acc, off, 32);
    if (lane == 0) out[e] = acc + b2[0];
  }
}

// ---------------------------------------------------------------------------
// Host orchestration
// ---------------------------------------------------------------------------
static inline unsigned blocks1d(size_t n) {
  size_t b = (n + THREADS - 1) / THREADS;
  if (b > (1u << 20)) b = 1u << 20;
  if (b == 0) b = 1;
  return (unsigned)b;
}

extern "C" void kernel_launch(void* const* d_in, const int* in_sizes, int n_in,
                              void* d_out, int out_size, void* d_ws, size_t ws_size,
                              hipStream_t stream) {
  const float* x    = (const float*)d_in[0];
  const int*   src  = (const int*)d_in[1];
  const int*   dst  = (const int*)d_in[2];
  const float* nw0  = (const float*)d_in[3];
  const float* nb0  = (const float*)d_in[4];
  const float* nw1  = (const float*)d_in[5];
  const float* nb1  = (const float*)d_in[6];
  const float* g1ws = (const float*)d_in[7];
  const float* g1bs = (const float*)d_in[8];
  const float* g1wd = (const float*)d_in[9];
  const float* g1bd = (const float*)d_in[10];
  const float* g1a  = (const float*)d_in[11];
  const float* g2ws = (const float*)d_in[12];
  const float* g2bs = (const float*)d_in[13];
  const float* g2wd = (const float*)d_in[14];
  const float* g2bd = (const float*)d_in[15];
  const float* g2a  = (const float*)d_in[16];
  const float* ew0  = (const float*)d_in[17];
  const float* eb0  = (const float*)d_in[18];
  const float* ew1  = (const float*)d_in[19];
  const float* eb1  = (const float*)d_in[20];
  const float* ew2  = (const float*)d_in[21];
  const float* eb2  = (const float*)d_in[22];

  const int IN_F = 64, H = 2, D = 128, HD = 256, NM = 128, EIN = 768, EH0 = 512, EH1 = 256;
  const int NN = in_sizes[0] / IN_F;  // 20000
  const int E  = in_sizes[1];         // 320000

  // --- workspace carve-out ---
  char* wp = (char*)d_ws;
  auto alloc = [&](size_t bytes) -> void* {
    void* p = (void*)wp;
    wp += (bytes + 255) & ~(size_t)255;
    return p;
  };
  _Float16* xh    = (_Float16*)alloc((size_t)NN * IN_F * 2);
  _Float16* nw0T  = (_Float16*)alloc((size_t)NM * IN_F * 2);
  _Float16* nw1T  = (_Float16*)alloc((size_t)NM * NM * 2);
  _Float16* g1wsT = (_Float16*)alloc((size_t)HD * IN_F * 2);
  _Float16* g1wdT = (_Float16*)alloc((size_t)HD * IN_F * 2);
  _Float16* g2wsT = (_Float16*)alloc((size_t)HD * HD * 2);
  _Float16* g2wdT = (_Float16*)alloc((size_t)HD * HD * 2);
  _Float16* ew0T  = (_Float16*)alloc((size_t)EH0 * EIN * 2);
  _Float16* ew1T  = (_Float16*)alloc((size_t)EH1 * EH0 * 2);
  _Float16* h1a   = (_Float16*)alloc((size_t)NN * NM * 2);
  _Float16* h1    = (_Float16*)alloc((size_t)NN * NM * 2);
  _Float16* fsb   = (_Float16*)alloc((size_t)NN * HD * 2);  // reused per layer
  _Float16* fdb   = (_Float16*)alloc((size_t)NN * HD * 2);
  float*    logit = (float*)alloc((size_t)E * H * 4);
  unsigned* maxk  = (unsigned*)alloc((size_t)NN * H * 4);
  float*    sums  = (float*)alloc((size_t)NN * H * 4);
  float*    agg   = (float*)alloc((size_t)NN * HD * 4);
  _Float16* h2a   = (_Float16*)alloc((size_t)NN * HD * 2);  // layer-1 output
  _Float16* h2    = (_Float16*)alloc((size_t)NN * HD * 2);  // layer-2 output
  _Float16* F     = (_Float16*)alloc((size_t)E * EIN * 2);
  _Float16* e0b   = (_Float16*)alloc((size_t)E * EH0 * 2);
  _Float16* e1b   = (_Float16*)alloc((size_t)E * EH1 * 2);

  auto gemm = [&](const _Float16* A, const _Float16* BT, const float* bias,
                  _Float16* C, int M, int N, int K, int relu) {
    dim3 g((M + 63) / 64, N / 128);
    k_gemm_f16<<<g, THREADS, 0, stream>>>(A, BT, bias, C, M, N, K, relu);
  };

  // --- precision conversion / weight transpose ---
  k_f32_to_f16<<<blocks1d((size_t)NN * IN_F), THREADS, 0, stream>>>(x, xh, (size_t)NN * IN_F);
  k_transpose_to_f16<<<blocks1d((size_t)IN_F * NM), THREADS, 0, stream>>>(nw0, nw0T, IN_F, NM);
  k_transpose_to_f16<<<blocks1d((size_t)NM * NM), THREADS, 0, stream>>>(nw1, nw1T, NM, NM);
  k_transpose_to_f16<<<blocks1d((size_t)IN_F * HD), THREADS, 0, stream>>>(g1ws, g1wsT, IN_F, HD);
  k_transpose_to_f16<<<blocks1d((size_t)IN_F * HD), THREADS, 0, stream>>>(g1wd, g1wdT, IN_F, HD);
  k_transpose_to_f16<<<blocks1d((size_t)HD * HD), THREADS, 0, stream>>>(g2ws, g2wsT, HD, HD);
  k_transpose_to_f16<<<blocks1d((size_t)HD * HD), THREADS, 0, stream>>>(g2wd, g2wdT, HD, HD);
  k_transpose_to_f16<<<blocks1d((size_t)EIN * EH0), THREADS, 0, stream>>>(ew0, ew0T, EIN, EH0);
  k_transpose_to_f16<<<blocks1d((size_t)EH0 * EH1), THREADS, 0, stream>>>(ew1, ew1T, EH0, EH1);

  // --- NodeMLP ---
  gemm(xh, nw0T, nb0, h1a, NN, NM, IN_F, 1);
  gemm(h1a, nw1T, nb1, h1, NN, NM, NM, 1);

  // --- GATv2 layer 1 (in: x f16) ---
  gemm(xh, g1wsT, g1bs, fsb, NN, HD, IN_F, 0);
  gemm(xh, g1wdT, g1bd, fdb, NN, HD, IN_F, 0);
  k_edge_logits<<<blocks1d((size_t)E * H), THREADS, 0, stream>>>(fsb, fdb, src, dst, g1a, logit, E, H, D);
  k_fill_u32<<<blocks1d((size_t)NN * H), THREADS, 0, stream>>>(maxk, 0u, (size_t)NN * H);
  k_seg_max<<<blocks1d((size_t)E * H), THREADS, 0, stream>>>(logit, dst, maxk, E, H);
  k_fill_f32<<<blocks1d((size_t)NN * H), THREADS, 0, stream>>>(sums, 0.0f, (size_t)NN * H);
  k_seg_expsum<<<blocks1d((size_t)E * H), THREADS, 0, stream>>>(logit, dst, maxk, sums, E, H);
  k_fill_f32<<<blocks1d((size_t)NN * HD), THREADS, 0, stream>>>(agg, 0.0f, (size_t)NN * HD);
  k_aggregate<<<blocks1d((size_t)E * H * (D / 4)), THREADS, 0, stream>>>(logit, sums, fsb, src, dst, agg, E, H, D);
  k_relu_f32_to_f16<<<blocks1d((size_t)NN * HD), THREADS, 0, stream>>>(agg, h2a, (size_t)NN * HD);

  // --- GATv2 layer 2 (in: h2a) ---
  gemm(h2a, g2wsT, g2bs, fsb, NN, HD, HD, 0);
  gemm(h2a, g2wdT, g2bd, fdb, NN, HD, HD, 0);
  k_edge_logits<<<blocks1d((size_t)E * H), THREADS, 0, stream>>>(fsb, fdb, src, dst, g2a, logit, E, H, D);
  k_fill_u32<<<blocks1d((size_t)NN * H), THREADS, 0, stream>>>(maxk, 0u, (size_t)NN * H);
  k_seg_max<<<blocks1d((size_t)E * H), THREADS, 0, stream>>>(logit, dst, maxk, E, H);
  k_fill_f32<<<blocks1d((size_t)NN * H), THREADS, 0, stream>>>(sums, 0.0f, (size_t)NN * H);
  k_seg_expsum<<<blocks1d((size_t)E * H), THREADS, 0, stream>>>(logit, dst, maxk, sums, E, H);
  k_fill_f32<<<blocks1d((size_t)NN * HD), THREADS, 0, stream>>>(agg, 0.0f, (size_t)NN * HD);
  k_aggregate<<<blocks1d((size_t)E * H * (D / 4)), THREADS, 0, stream>>>(logit, sums, fsb, src, dst, agg, E, H, D);
  k_relu_f32_to_f16<<<blocks1d((size_t)NN * HD), THREADS, 0, stream>>>(agg, h2, (size_t)NN * HD);

  // --- EdgeMLP (dominant ~336 GFLOP, all WMMA) ---
  k_gather_edges<<<blocks1d((size_t)E * 96), THREADS, 0, stream>>>(h1, h2, src, dst, F, E);
  // relu fused at producer == relu applied to consumer input (matches reference)
  gemm(F, ew0T, eb0, e0b, E, EH0, EIN, 1);
  gemm(e0b, ew1T, eb1, e1b, E, EH1, EH0, 1);
  k_final_dot<<<blocks1d((size_t)E * 32), THREADS, 0, stream>>>(e1b, ew2, eb2, (float*)d_out, E);
}
